// GINEMalwareClassifier_6880537608212
// MI455X (gfx1250) — matrix operand order repaired and checked by reading, not compiled, not run on hardware
//
#include <hip/hip_runtime.h>
#include <hip/hip_bf16.h>

#define N_NODES   50000
#define N_EDGES   640000
#define F_IN      64
#define EDGE_DIM  16
#define HID       128
#define N_LAYERS  3
#define N_CLASSES 2
#define NUM_GRAPHS 64
#define BN_EPS 1e-5f

typedef __attribute__((ext_vector_type(16))) _Float16 v16h;
typedef __attribute__((ext_vector_type(8)))  float    v8f;

union Frag {
    v16h v;
    _Float16 h[16];
    uint4 q[2];
};

// ---------------------------------------------------------------------------
// Generic tiled GEMM with optional elementwise A2 add, bias and ReLU:
//   C[M,N] = act((A (+ A2)) @ W + bias)
// Requirements honored by all call sites: K % 32 == 0; N == 128 or N < 128
// with a single column tile (uniform scalar fallback path).
// Block tile: 64 (M) x 128 (N), K-step 32, 4 waves (wave w owns rows 16w..16w+15).
// Staging is branchless (row/col clamped; garbage rows/cols are masked by the
// guarded epilogue). WMMA: v_wmma_f32_16x16x32_f16, fragments via ds_load_b128.
// ---------------------------------------------------------------------------
#define TM 64
#define TN 128
#define KS 32
#define LDA 40   // padded LDS row stride (halves); keeps 16B alignment

__global__ __launch_bounds__(128, 2)
void gemm_wmma_kernel(const float* __restrict__ A, const float* __restrict__ A2,
                      const float* __restrict__ W, const float* __restrict__ bias,
                      float* __restrict__ C, int M, int K, int N, int do_relu)
{
    __shared__ __align__(16) _Float16 As[TM][LDA];
    __shared__ __align__(16) _Float16 BsT[TN][LDA];

    const int tid  = threadIdx.x;
    const int wave = tid >> 5;
    const int lane = tid & 31;
    const int half = lane >> 4;   // 0 or 1 (half-wave)
    const int l15  = lane & 15;
    const int m0 = blockIdx.x * TM;
    const int n0 = blockIdx.y * TN;

    v8f acc[8];
    for (int t = 0; t < 8; ++t)
        for (int j = 0; j < 8; ++j) acc[t][j] = 0.0f;

    for (int k0 = 0; k0 < K; k0 += KS) {
        // ---- stage A tile: 64x32 fp32 -> f16, float4 loads, clamped rows ----
        for (int i = tid; i < (TM * KS) / 4; i += 128) {
            int r  = i >> 3;            // 8 float4 per row
            int c4 = (i & 7) << 2;
            int gm = m0 + r; if (gm >= M) gm = M - 1;   // clamp: value irrelevant
            const float* ap = A + (size_t)gm * K + (k0 + c4);
            float4 v = *(const float4*)ap;
            if (A2) {
                float4 v2 = *(const float4*)(A2 + (size_t)gm * K + (k0 + c4));
                v.x += v2.x; v.y += v2.y; v.z += v2.z; v.w += v2.w;
            }
            As[r][c4 + 0] = (_Float16)v.x;
            As[r][c4 + 1] = (_Float16)v.y;
            As[r][c4 + 2] = (_Float16)v.z;
            As[r][c4 + 3] = (_Float16)v.w;
            if (k0 + KS < K) __builtin_prefetch(ap + KS, 0, 1);
        }
        // ---- stage B tile transposed: BsT[n][k] ----
        if (N >= TN) {   // uniform branch; N == 128 here, fully in range
            for (int i = tid; i < (KS * TN) / 4; i += 128) {
                int kk = i >> 5;            // 32 float4 per k-row
                int c4 = (i & 31) << 2;
                const float* wp = W + (size_t)(k0 + kk) * N + (n0 + c4);
                float4 v = *(const float4*)wp;
                BsT[c4 + 0][kk] = (_Float16)v.x;
                BsT[c4 + 1][kk] = (_Float16)v.y;
                BsT[c4 + 2][kk] = (_Float16)v.z;
                BsT[c4 + 3][kk] = (_Float16)v.w;
                if (k0 + KS < K) __builtin_prefetch(wp + (size_t)KS * N, 0, 1);
            }
        } else {         // tiny-N fallback (head output, N=2)
            for (int i = tid; i < KS * TN; i += 128) {
                int kk = i >> 7, c = i & (TN - 1);
                int gn = n0 + c;
                float v = (gn < N) ? W[(size_t)(k0 + kk) * N + gn] : 0.0f;
                BsT[c][kk] = (_Float16)v;
            }
        }
        __syncthreads();

        // A fragment: lane holds row (wave*16 + l15); halves 0..7 = K kb..kb+7,
        // halves 8..15 = K kb+16..kb+23, kb = half*8 (ISA 16-bit A layout)
        Frag af;
        const int arow = wave * 16 + l15;
        const int akb  = half * 8;
        af.q[0] = *(const uint4*)&As[arow][akb];
        af.q[1] = *(const uint4*)&As[arow][akb + 16];

        const int bkb = half * 8;   // B: lane holds col l15, K = bkb..bkb+7, bkb+16..bkb+23
        #pragma unroll
        for (int t = 0; t < 8; ++t) {
            Frag bf;
            const int bcol = t * 16 + l15;
            bf.q[0] = *(const uint4*)&BsT[bcol][bkb];
            bf.q[1] = *(const uint4*)&BsT[bcol][bkb + 16];
            acc[t] = __builtin_amdgcn_wmma_f32_16x16x32_f16(
                false, af.v, false, bf.v, (short)0, acc[t], false, false);
        }
        __syncthreads();
    }

    // epilogue: C/D layout -> lane covers col l15, VGPR j covers row j (+8 for half 1)
    for (int t = 0; t < 8; ++t) {
        int n = n0 + t * 16 + l15;
        if (n >= N) continue;
        float bv = bias ? bias[n] : 0.0f;
        #pragma unroll
        for (int j = 0; j < 8; ++j) {
            int m = m0 + wave * 16 + half * 8 + j;
            if (m < M) {
                float v = acc[t][j] + bv;
                if (do_relu) v = fmaxf(v, 0.0f);
                C[(size_t)m * N + n] = v;
            }
        }
    }
}

// ---------------------------------------------------------------------------
// Fused GINE edge phase: for each edge, msg = ReLU(h[src] + edge_attr @ eW + eb)
// scatter-added into agg[dst]. 64 edges per block (16 per wave). The edge
// linear runs on WMMA (K padded 16->32; elements 8..15 of each lane carry zeros).
// ---------------------------------------------------------------------------
#define EB 64

__global__ __launch_bounds__(128, 2)
void gine_edge_kernel(const float* __restrict__ h,
                      const int* __restrict__ src, const int* __restrict__ dst,
                      const float* __restrict__ edge_attr,
                      const float* __restrict__ eW, const float* __restrict__ eb,
                      float* __restrict__ agg)
{
    __shared__ int s_src[EB], s_dst[EB];
    __shared__ __align__(16) float    s_h[EB][HID];      // gathered h[src] rows
    __shared__ __align__(16) _Float16 s_wT[HID][16];     // eW transposed, f16

    const int tid  = threadIdx.x;
    const int wave = tid >> 5;
    const int lane = tid & 31;
    const int half = lane >> 4;
    const int l15  = lane & 15;
    const int e0   = blockIdx.x * EB;

    if (tid < EB) {
        int e = e0 + tid;
        int ok = (e < N_EDGES);
        s_src[tid] = ok ? src[e] : 0;
        s_dst[tid] = ok ? dst[e] : 0;
    }
    for (int i = tid; i < EDGE_DIM * HID; i += 128) {
        int k = i >> 7, n = i & (HID - 1);
        s_wT[n][k] = (_Float16)eW[i];
    }
    __syncthreads();

    // cooperatively gather the 16 h[src] rows owned by this wave (float4 loads)
    for (int r = 0; r < 16; ++r) {
        int row = wave * 16 + r;
        const float4* hp = (const float4*)(h + (size_t)s_src[row] * HID);
        ((float4*)s_h[row])[lane] = hp[lane];
    }
    __syncthreads();

    // A fragment: 16 edges x K32 (edge_attr dims 0..15, rest zero)
    Frag af;
    {
        int e  = e0 + wave * 16 + l15;
        int ee = (e < N_EDGES) ? e : 0;
        const float* ap = edge_attr + (size_t)ee * EDGE_DIM;
        int kb = half * 8;
        float4 a0 = *(const float4*)(ap + kb);
        float4 a1 = *(const float4*)(ap + kb + 4);
        af.h[0] = (_Float16)a0.x; af.h[1] = (_Float16)a0.y;
        af.h[2] = (_Float16)a0.z; af.h[3] = (_Float16)a0.w;
        af.h[4] = (_Float16)a1.x; af.h[5] = (_Float16)a1.y;
        af.h[6] = (_Float16)a1.z; af.h[7] = (_Float16)a1.w;
        #pragma unroll
        for (int i = 0; i < 8; ++i) af.h[8 + i] = (_Float16)0.0f;  // K pad
    }

    #pragma unroll
    for (int t = 0; t < 8; ++t) {
        Frag bf;
        int col = t * 16 + l15;
        // s0..7: K = half*8..half*8+7 real weights; s8..15: K 16..31 zero padding
        bf.q[0] = *(const uint4*)&s_wT[col][half * 8];
        bf.q[1] = make_uint4(0u, 0u, 0u, 0u);
        v8f out;
        for (int j = 0; j < 8; ++j) out[j] = 0.0f;
        out = __builtin_amdgcn_wmma_f32_16x16x32_f16(
            false, af.v, false, bf.v, (short)0, out, false, false);

        float bv = eb[col];
        #pragma unroll
        for (int j = 0; j < 8; ++j) {
            int mrow = wave * 16 + half * 8 + j;
            int e = e0 + mrow;
            if (e < N_EDGES) {
                float v = out[j] + s_h[mrow][col] + bv;
                v = fmaxf(v, 0.0f);
                atomicAdd(&agg[(size_t)s_dst[mrow] * HID + col], v);
            }
        }
    }
}

// ---------------------------------------------------------------------------
// Small helper kernels
// ---------------------------------------------------------------------------
__global__ void zero_kernel(float* __restrict__ p, long n) {
    long i = (long)blockIdx.x * blockDim.x + threadIdx.x;
    long stride = (long)gridDim.x * blockDim.x;
    for (; i < n; i += stride) p[i] = 0.0f;
}

#define BN_ROWS 512
__global__ __launch_bounds__(HID)
void bn_stats_kernel(const float* __restrict__ z, float* __restrict__ sums, int M) {
    int c  = threadIdx.x;                  // one channel per thread (coalesced rows)
    int r0 = blockIdx.x * BN_ROWS;
    int r1 = r0 + BN_ROWS; if (r1 > M) r1 = M;
    float s = 0.0f, s2 = 0.0f;
    for (int r = r0; r < r1; ++r) {
        float v = z[(size_t)r * HID + c];
        s += v; s2 += v * v;
    }
    atomicAdd(&sums[c], s);
    atomicAdd(&sums[HID + c], s2);
}

__global__ void bn_apply_kernel(const float* __restrict__ z, const float* __restrict__ sums,
                                const float* __restrict__ gamma, const float* __restrict__ beta,
                                float* __restrict__ out, int M) {
    int total = M * HID;
    float inv = 1.0f / (float)M;
    for (int i = blockIdx.x * blockDim.x + threadIdx.x; i < total;
         i += gridDim.x * blockDim.x) {
        int c = i & (HID - 1);
        float mean = sums[c] * inv;
        float var  = sums[HID + c] * inv - mean * mean;
        float v = gamma[c] * (z[i] - mean) * rsqrtf(var + BN_EPS) + beta[c];
        out[i] = fmaxf(v, 0.0f);
    }
}

__global__ void pool_kernel(const float* __restrict__ h, const int* __restrict__ batch,
                            float* __restrict__ g) {
    int total = N_NODES * HID;
    for (int i = blockIdx.x * blockDim.x + threadIdx.x; i < total;
         i += gridDim.x * blockDim.x) {
        int r = i >> 7, c = i & (HID - 1);
        atomicAdd(&g[(size_t)batch[r] * HID + c], h[i]);
    }
}

// ---------------------------------------------------------------------------
extern "C" void kernel_launch(void* const* d_in, const int* in_sizes, int n_in,
                              void* d_out, int out_size, void* d_ws, size_t ws_size,
                              hipStream_t stream)
{
    const float* x        = (const float*)d_in[0];
    const int*   eidx     = (const int*)  d_in[1];
    const float* eattr    = (const float*)d_in[2];
    const int*   batch    = (const int*)  d_in[3];
    const float* enc_W    = (const float*)d_in[4];
    const float* enc_b    = (const float*)d_in[5];
    const float* edge_W   = (const float*)d_in[6];
    const float* edge_b   = (const float*)d_in[7];
    const float* mlp_W1   = (const float*)d_in[8];
    const float* mlp_b1   = (const float*)d_in[9];
    const float* mlp_W2   = (const float*)d_in[10];
    const float* mlp_b2   = (const float*)d_in[11];
    const float* bn_gamma = (const float*)d_in[12];
    const float* bn_beta  = (const float*)d_in[13];
    const float* head_W1  = (const float*)d_in[14];
    const float* head_b1  = (const float*)d_in[15];
    const float* head_W2  = (const float*)d_in[16];
    const float* head_b2  = (const float*)d_in[17];

    const int* esrc = eidx;              // edge_index[0]
    const int* edst = eidx + N_EDGES;    // edge_index[1]

    const size_t NH = (size_t)N_NODES * HID;
    float* h    = (float*)d_ws;          // node features
    float* agg  = h + NH;                // scatter target / mlp2 out
    float* buf1 = agg + NH;              // mlp1 out
    float* g    = buf1 + NH;             // pooled [64,128]
    float* gh   = g + (size_t)NUM_GRAPHS * HID;
    float* sums = gh + (size_t)NUM_GRAPHS * HID;  // BN sum/sumsq [2*128]

    dim3 blk(128);
    dim3 node_grid((N_NODES + TM - 1) / TM, 1);

    // node encoder: h = x @ enc_W + enc_b
    gemm_wmma_kernel<<<node_grid, blk, 0, stream>>>(
        x, nullptr, enc_W, enc_b, h, N_NODES, F_IN, HID, 0);

    for (int l = 0; l < N_LAYERS; ++l) {
        zero_kernel<<<2048, 256, 0, stream>>>(agg, (long)NH);
        gine_edge_kernel<<<(N_EDGES + EB - 1) / EB, blk, 0, stream>>>(
            h, esrc, edst, eattr,
            edge_W + (size_t)l * EDGE_DIM * HID, edge_b + l * HID, agg);
        // buf1 = ReLU((h + agg) @ W1 + b1)   (A2 fuses z = h + agg)
        gemm_wmma_kernel<<<node_grid, blk, 0, stream>>>(
            h, agg, mlp_W1 + (size_t)l * HID * HID, mlp_b1 + l * HID, buf1,
            N_NODES, HID, HID, 1);
        // agg = buf1 @ W2 + b2
        gemm_wmma_kernel<<<node_grid, blk, 0, stream>>>(
            buf1, nullptr, mlp_W2 + (size_t)l * HID * HID, mlp_b2 + l * HID, agg,
            N_NODES, HID, HID, 0);
        // BatchNorm (batch stats) + ReLU -> h
        zero_kernel<<<1, 256, 0, stream>>>(sums, 2 * HID);
        bn_stats_kernel<<<(N_NODES + BN_ROWS - 1) / BN_ROWS, HID, 0, stream>>>(
            agg, sums, N_NODES);
        bn_apply_kernel<<<1024, 256, 0, stream>>>(
            agg, sums, bn_gamma + l * HID, bn_beta + l * HID, h, N_NODES);
    }

    // global_add_pool + head
    zero_kernel<<<8, 256, 0, stream>>>(g, (long)NUM_GRAPHS * HID);
    pool_kernel<<<2048, 256, 0, stream>>>(h, batch, g);
    gemm_wmma_kernel<<<dim3(1, 1), blk, 0, stream>>>(
        g, nullptr, head_W1, head_b1, gh, NUM_GRAPHS, HID, HID, 1);
    gemm_wmma_kernel<<<dim3(1, 1), blk, 0, stream>>>(
        gh, nullptr, head_W2, head_b2, (float*)d_out, NUM_GRAPHS, HID, N_CLASSES, 0);
}
